// MEL_67035849556474
// MI455X (gfx1250) — matrix operand order repaired and checked by reading, not compiled, not run on hardware
//
#include <hip/hip_runtime.h>
#include <hip/hip_bf16.h>
#include <math.h>

typedef __attribute__((ext_vector_type(16))) _Float16 v16h;
typedef __attribute__((ext_vector_type(8)))  _Float16 v8h;
typedef __attribute__((ext_vector_type(8)))  float    v8f;

#define GAMMA_   20.0f
#define GAMMA2_  10.0f
#define EPSN     1e-8f

// Problem constants (match setup_inputs: b=16, n_way=5, k_shot=5, q=75, c=640, h=w=5)
#define B_    16
#define NWAY  5
#define KSHOT 5
#define Q_    75
#define C_    640
#define HW    25
#define MS    125           // n_way*hw
#define NB    1200          // B_*Q_

// ---------------------------------------------------------------------------
// Kernel 0: zero the scalar output (harness poisons d_out; must re-zero per call)
// ---------------------------------------------------------------------------
__global__ void mel_zero(float* out) { if (threadIdx.x == 0) out[0] = 0.0f; }

// ---------------------------------------------------------------------------
// Kernel 1: support prep. mean over k_shot, L2-normalize over c, write f16
// P[b][n=128][k=640] row-major (n = w*25+j; rows 125..127 zero padding).
// Row-major-[n][k] is exactly the B^T layout the WMMA B fragment wants.
// ---------------------------------------------------------------------------
__global__ void mel_prep_support(const float* __restrict__ sx, _Float16* __restrict__ P) {
    __shared__ float red[256];
    __shared__ float s_inv;
    const int blk = blockIdx.x;            // b*128 + nj
    const int b   = blk >> 7;
    const int nj  = blk & 127;
    const int t   = threadIdx.x;
    _Float16* prow = P + (size_t)blk * C_;
    if (nj >= MS) {                         // padding rows -> zeros
        for (int c = t; c < C_; c += 256) prow[c] = (_Float16)0.0f;
        return;
    }
    const int w = nj / HW, j = nj % HW;
    const int c0 = t, c1 = t + 256, c2 = t + 512;   // c0,c1 always valid; c2 iff t<128
    float m0 = 0.f, m1 = 0.f, m2 = 0.f;
    {
        const float* base = sx + ((size_t)(b * (NWAY*KSHOT) + w * KSHOT) * C_) * HW + j;
        const size_t sstr = (size_t)C_ * HW;
        for (int s = 0; s < KSHOT; ++s) {
            const float* p = base + (size_t)s * sstr;
            m0 += p[(size_t)c0 * HW];
            m1 += p[(size_t)c1 * HW];
            if (c2 < C_) m2 += p[(size_t)c2 * HW];
        }
        m0 *= 0.2f; m1 *= 0.2f; m2 *= 0.2f;
    }
    red[t] = m0*m0 + m1*m1 + m2*m2;
    __syncthreads();
    for (int o = 128; o > 0; o >>= 1) { if (t < o) red[t] += red[t + o]; __syncthreads(); }
    if (t == 0) s_inv = 1.0f / (sqrtf(red[0]) + EPSN);
    __syncthreads();
    const float inv = s_inv;
    prow[c0] = (_Float16)(m0 * inv);
    prow[c1] = (_Float16)(m1 * inv);
    if (c2 < C_) prow[c2] = (_Float16)(m2 * inv);
}

// ---------------------------------------------------------------------------
// Kernel 2: query prep. L2-normalize over c per (b,q,i); write f16
// Qt[n][i=32][k=640] row-major (rows 25..31 zero) = A-matrix layout.
// ---------------------------------------------------------------------------
__global__ void mel_prep_query(const float* __restrict__ qx, _Float16* __restrict__ Qt) {
    __shared__ float red[256];
    __shared__ float s_inv;
    const int n = blockIdx.x;              // 0..1199 = b*75+q
    const int t = threadIdx.x;
    const float* qbase = qx + (size_t)n * C_ * HW;
    _Float16* obase = Qt + (size_t)n * 32 * C_;
    const int c0 = t, c1 = t + 256, c2 = t + 512;
    for (int i = 0; i < HW; ++i) {
        float m0 = qbase[(size_t)c0 * HW + i];
        float m1 = qbase[(size_t)c1 * HW + i];
        float m2 = (c2 < C_) ? qbase[(size_t)c2 * HW + i] : 0.0f;
        red[t] = m0*m0 + m1*m1 + m2*m2;
        __syncthreads();
        for (int o = 128; o > 0; o >>= 1) { if (t < o) red[t] += red[t + o]; __syncthreads(); }
        if (t == 0) s_inv = 1.0f / (sqrtf(red[0]) + EPSN);
        __syncthreads();
        const float inv = s_inv;
        _Float16* orow = obase + (size_t)i * C_;
        orow[c0] = (_Float16)(m0 * inv);
        orow[c1] = (_Float16)(m1 * inv);
        if (c2 < C_) orow[c2] = (_Float16)(m2 * inv);
        __syncthreads();                   // red / s_inv reused next iteration
    }
    for (int i = HW; i < 32; ++i) {        // zero pad rows so WMMA A-frags are clean
        _Float16* orow = obase + (size_t)i * C_;
        for (int c = t; c < C_; c += 256) orow[c] = (_Float16)0.0f;
    }
}

// ---------------------------------------------------------------------------
// Kernel 3: fused per-(b,q) pipeline.
//   S(25x125) = Qn x Pn^T via v_wmma_f32_16x16x32_f16 (K=640)
//   U = T_qs (softmax over i, gamma2),  V = T_sq (softmax over wj, gamma) in-place
//   Schur: (I - 0.25*C*B) x = 1 + 0.5*C*1  (25x25 Gaussian elimination)
//   katz_s = 0.5*B*x ; predicts = group sums ; loss += (log(tot)-log(pred[y]))/N
// ---------------------------------------------------------------------------
__global__ void __launch_bounds__(256, 2)
mel_main(const _Float16* __restrict__ Qt, const _Float16* __restrict__ P,
         const int* __restrict__ qy, float* __restrict__ out) {
    __shared__ float S[32][128];   // 16 KB: raw S, then V=T_sq in rows 0..24
    __shared__ float U[125][28];   // 14 KB: U[wj][i] = T_qs[wj,i]
    __shared__ float A[25][26];    // augmented 25x25 system
    __shared__ float xs[25];
    __shared__ float ks[125];
    __shared__ float pred[NWAY];

    const int n    = blockIdx.x;
    const int b    = n / Q_;
    const int tid  = threadIdx.x;
    const int wave = tid >> 5;
    const int lane = tid & 31;

    // ---------------- WMMA GEMM: 2 M-tiles x 8 N-tiles, each wave owns 2 N-tiles
    {
        const _Float16* Abase = Qt + (size_t)n * 32 * C_;
        const _Float16* Bbase = P  + (size_t)b * 128 * C_;
        const int mt  = wave >> 2;            // 0..1
        const int nt0 = (wave & 3) * 2;       // 0,2,4,6
        const int kh  = lane >> 4;            // half-wave selector
        const _Float16* arow  = Abase + (size_t)(mt * 16 + (lane & 15)) * C_;
        const _Float16* brow0 = Bbase + (size_t)(nt0 * 16 + (lane & 15)) * C_;
        const _Float16* brow1 = brow0 + (size_t)16 * C_;

        v8f acc0 = {}; v8f acc1 = {};
        for (int kk = 0; kk < C_; kk += 32) {
            v16h af, bf0, bf1;
            {   // A frag: lanes<16 hold K kk..kk+15; lanes>=16 kk+16..kk+31
                v8h lo = *(const v8h*)(arow + kk + kh * 16);
                v8h hi = *(const v8h*)(arow + kk + kh * 16 + 8);
                for (int u = 0; u < 8; ++u) { af[u] = lo[u]; af[8 + u] = hi[u]; }
            }
            {   // B frag (B^T row-major): lanes<16 hold K kk..kk+15; lanes>=16 kk+16..kk+31
                const v8h* p = (const v8h*)(brow0 + kk + kh * 16);
                v8h lo = p[0], hi = p[1];
                for (int u = 0; u < 8; ++u) { bf0[u] = lo[u]; bf0[8 + u] = hi[u]; }
            }
            {
                const v8h* p = (const v8h*)(brow1 + kk + kh * 16);
                v8h lo = p[0], hi = p[1];
                for (int u = 0; u < 8; ++u) { bf1[u] = lo[u]; bf1[8 + u] = hi[u]; }
            }
            acc0 = __builtin_amdgcn_wmma_f32_16x16x32_f16(false, af, false, bf0, (short)0, acc0, false, false);
            acc1 = __builtin_amdgcn_wmma_f32_16x16x32_f16(false, af, false, bf1, (short)0, acc1, false, false);
        }
        // C/D layout: VGPR r -> row mt*16 + r + 8*(lane>=16), col = nt*16 + lane%16
        const int rbase = mt * 16 + 8 * kh;
        const int col   = lane & 15;
        for (int r = 0; r < 8; ++r) {
            S[rbase + r][nt0 * 16 + col]       = acc0[r];
            S[rbase + r][(nt0 + 1) * 16 + col] = acc1[r];
        }
    }
    __syncthreads();

    // ---------------- U = T_qs: softmax over i (25) of gamma2*S[i][wj], per column wj
    if (tid < MS) {
        const int wj = tid;
        float mx = -3.0e38f;
        for (int i = 0; i < HW; ++i) mx = fmaxf(mx, S[i][wj]);
        mx *= GAMMA2_;
        float sm = 0.f;
        for (int i = 0; i < HW; ++i) sm += __expf(GAMMA2_ * S[i][wj] - mx);
        const float inv = 1.0f / sm;
        for (int i = 0; i < HW; ++i) U[wj][i] = __expf(GAMMA2_ * S[i][wj] - mx) * inv;
    }
    __syncthreads();   // U must be done before S is overwritten by V

    // ---------------- V = T_sq in place: row softmax over wj (125) of gamma*S[i][:]
    for (int i = wave; i < HW; i += 8) {
        float mx = -3.0e38f;
        for (int c = lane; c < MS; c += 32) mx = fmaxf(mx, S[i][c]);
        for (int o = 16; o > 0; o >>= 1) mx = fmaxf(mx, __shfl_xor(mx, o));
        mx *= GAMMA_;
        float sm = 0.f;
        for (int c = lane; c < MS; c += 32) { float e = __expf(GAMMA_ * S[i][c] - mx); S[i][c] = e; sm += e; }
        for (int o = 16; o > 0; o >>= 1) sm += __shfl_xor(sm, o);
        const float inv = 1.0f / sm;
        for (int c = lane; c < MS; c += 32) S[i][c] *= inv;
    }
    __syncthreads();

    // ---------------- A = I - 0.25 * (C B) ; rhs = 1 + 0.5 * C*1  (col 25)
    // (CB)[i][j] = sum_wj U[wj][i] * V[j][wj]
    for (int t = tid; t < 25 * 26; t += 256) {
        const int i = t / 26, j = t % 26;
        if (j < 25) {
            float s = 0.f;
            for (int wj = 0; wj < MS; ++wj) s += U[wj][i] * S[j][wj];
            A[i][j] = ((i == j) ? 1.0f : 0.0f) - 0.25f * s;
        } else {
            float s = 0.f;
            for (int wj = 0; wj < MS; ++wj) s += U[wj][i];
            A[i][25] = 1.0f + 0.5f * s;
        }
    }
    __syncthreads();

    // ---------------- Gaussian elimination (no pivot; A is a Katz M-matrix) + back-sub
    for (int k = 0; k < 24; ++k) {
        if (tid > k && tid < 25) {
            const float f = A[tid][k] / A[k][k];
            for (int j = k; j < 26; ++j) A[tid][j] -= f * A[k][j];
        }
        __syncthreads();
    }
    for (int k = 24; k >= 0; --k) {
        if (tid == 0) xs[k] = A[k][25] / A[k][k];
        __syncthreads();
        if (tid < k) A[tid][25] -= A[tid][k] * xs[k];
        __syncthreads();
    }

    // ---------------- katz_s = 0.5 * B x  (B[wj][i] = V[i][wj] = S[i][wj])
    if (tid < MS) {
        float s = 0.f;
        for (int i = 0; i < HW; ++i) s += S[i][tid] * xs[i];
        ks[tid] = 0.5f * s;
    }
    __syncthreads();
    if (tid < NWAY) {
        float s = 0.f;
        for (int j = 0; j < HW; ++j) s += ks[tid * HW + j];
        pred[tid] = s;
    }
    __syncthreads();
    if (tid == 0) {
        const float tot = pred[0] + pred[1] + pred[2] + pred[3] + pred[4];
        const int y = qy[n];
        const float l = (logf(tot) - logf(pred[y])) * (1.0f / (float)NB);
        atomicAdd(out, l);
    }
}

// ---------------------------------------------------------------------------
// Launch. d_in order: support_xf, support_y, query_xf, query_y, n_way, k_shot.
// Workspace: P = 16*128*640 f16 (2.6 MB), Qt = 1200*32*640 f16 (49.2 MB).
// ---------------------------------------------------------------------------
extern "C" void kernel_launch(void* const* d_in, const int* in_sizes, int n_in,
                              void* d_out, int out_size, void* d_ws, size_t ws_size,
                              hipStream_t stream) {
    const float* sx = (const float*)d_in[0];
    const float* qx = (const float*)d_in[2];
    const int*   qy = (const int*)d_in[3];
    float* out = (float*)d_out;

    _Float16* P  = (_Float16*)d_ws;
    _Float16* Qt = P + (size_t)B_ * 128 * C_;   // needs ~51.8 MB of d_ws total

    mel_zero<<<1, 32, 0, stream>>>(out);
    mel_prep_support<<<B_ * 128, 256, 0, stream>>>(sx, P);
    mel_prep_query<<<NB, 256, 0, stream>>>(qx, Qt);
    mel_main<<<NB, 256, 0, stream>>>(Qt, P, qy, out);
}
